// DynamicBasisQueryAttention_15144054685912
// MI455X (gfx1250) — matrix operand order repaired and checked by reading, not compiled, not run on hardware
//
#include <hip/hip_runtime.h>
#include <cstdint>
#include <cstddef>

typedef __attribute__((ext_vector_type(16))) _Float16 v16h;
typedef __attribute__((ext_vector_type(8)))  _Float16 v8h;
typedef __attribute__((ext_vector_type(8)))  float    v8f;
typedef __attribute__((ext_vector_type(4)))  unsigned int u32x4;
typedef __attribute__((ext_vector_type(8)))  int      i32x8;
typedef __attribute__((ext_vector_type(4)))  int      i32x4;

#define WMMA_F16(a, b, c) \
  __builtin_amdgcn_wmma_f32_16x16x32_f16(false, (a), false, (b), (short)0, (c), false, false)

// Use Tensor Data Mover for the attention K-tile loads (6-arg builtin variant:
// amdgpu-toolchain clang-23 / therock-10.0 headers).
#define USE_TDM 1

// Problem constants
constexpr int kB  = 2;
constexpr int kT  = 2048;
constexpr int kC  = 1024;
constexpr int kH  = 16;
constexpr int kJ  = 4;
constexpr int kD  = 64;
constexpr int kJD = 256;             // J*D: effective head dim
constexpr int kM  = kB * kT;         // 4096 rows

// ---------------------------------------------------------------------------
// WMMA fragment loaders (wave32, 16x16x32 f16 shapes per CDNA5 ISA 7.12.2)
// ---------------------------------------------------------------------------
__device__ __forceinline__ v16h load_frag_a(const _Float16* p, int stride) {
  const int lane = threadIdx.x & 31;
  const int m  = lane & 15;
  const int hi = (lane >> 4) & 1;
  v16h a;
#pragma unroll
  for (int i = 0; i < 16; ++i) {
    const int k = i + 16 * hi;
    a[i] = p[m * stride + k];
  }
  return a;
}

__device__ __forceinline__ v16h load_frag_b(const _Float16* p, int kstride, int nstride) {
  const int lane = threadIdx.x & 31;
  const int n  = lane & 15;
  const int hi = (lane >> 4) & 1;
  v16h b;
#pragma unroll
  for (int i = 0; i < 16; ++i) {
    const int k = i + 16 * hi;
    b[i] = p[k * kstride + n * nstride];
  }
  return b;
}

// ---------------------------------------------------------------------------
// fp32 -> f16 convert
// ---------------------------------------------------------------------------
__global__ void cvt_f32_f16(const float* __restrict__ src, _Float16* __restrict__ dst, int n) {
  int i = blockIdx.x * blockDim.x + threadIdx.x;
  if (i < n) dst[i] = (_Float16)src[i];
}

// ---------------------------------------------------------------------------
// Generic GEMM: Out[m,n] = sum_k A[m,k] * W[n,k]   (A: MxK f16, W: NxK f16)
// Block: 256 threads (8 waves), 64x64 output tile, BK=64 (4 WMMAs/wave/sync).
// ---------------------------------------------------------------------------
__global__ void gemm_f16f32(const _Float16* __restrict__ A, const _Float16* __restrict__ W,
                            float* __restrict__ Out, int M, int N, int K) {
  __shared__ _Float16 As[64][72];
  __shared__ _Float16 Bs[64][72];
  const int m0 = blockIdx.x * 64, n0 = blockIdx.y * 64;
  const int tid = threadIdx.x, wave = tid >> 5, lane = tid & 31;
  const int tm = (wave & 3) * 16, tn = (wave >> 2) * 16;
  const int lr = tid >> 2, lc = (tid & 3) * 16;
  v8f acc0 = {};
  v8f acc1 = {};
  for (int k0 = 0; k0 < K; k0 += 64) {
    const _Float16* ga = A + (size_t)(m0 + lr) * K + k0 + lc;
    const _Float16* gb = W + (size_t)(n0 + lr) * K + k0 + lc;
    *(v8h*)&As[lr][lc]     = *(const v8h*)ga;
    *(v8h*)&As[lr][lc + 8] = *(const v8h*)(ga + 8);
    *(v8h*)&Bs[lr][lc]     = *(const v8h*)gb;
    *(v8h*)&Bs[lr][lc + 8] = *(const v8h*)(gb + 8);
    __syncthreads();
    v16h a0  = load_frag_a(&As[tm][0], 72);
    v16h a1  = load_frag_a(&As[tm][32], 72);
    v16h b00 = load_frag_b(&Bs[tn][0], 1, 72);
    v16h b01 = load_frag_b(&Bs[tn][32], 1, 72);
    v16h b10 = load_frag_b(&Bs[tn + 32][0], 1, 72);
    v16h b11 = load_frag_b(&Bs[tn + 32][32], 1, 72);
    acc0 = WMMA_F16(a0, b00, acc0);
    acc0 = WMMA_F16(a1, b01, acc0);
    acc1 = WMMA_F16(a0, b10, acc1);
    acc1 = WMMA_F16(a1, b11, acc1);
    __syncthreads();
  }
  const int n = lane & 15, hi = lane >> 4;
#pragma unroll
  for (int r = 0; r < 8; ++r) {
    const int m = m0 + tm + hi * 8 + r;
    Out[(size_t)m * N + n0 + tn + n]      = acc0[r];
    Out[(size_t)m * N + n0 + tn + 32 + n] = acc1[r];
  }
}

// ---------------------------------------------------------------------------
// Prep: rotary + rms_norm on q/k, softmax(w_k/w_v), gate, build q_eff/k_flat.
// ---------------------------------------------------------------------------
__global__ void prep_kernel(const float* __restrict__ qraw, const float* __restrict__ kraw,
                            const float* __restrict__ vraw, const float* __restrict__ akraw,
                            const float* __restrict__ avraw, const float* __restrict__ x,
                            const float* __restrict__ cosb, const float* __restrict__ sinb,
                            const float* __restrict__ bak, const float* __restrict__ bav,
                            const float* __restrict__ wgate,
                            _Float16* __restrict__ qeff, _Float16* __restrict__ kflat,
                            _Float16* __restrict__ vbh, float* __restrict__ wvw,
                            float* __restrict__ gateg) {
  __shared__ float qn[kH][kD];
  __shared__ float wks[kH][kJ];
  const int row = blockIdx.x;
  const int b = row / kT, t = row % kT;
  const int tid = threadIdx.x, wave = tid >> 5, lane = tid & 31;
  const float cv = cosb[t * 32 + lane];
  const float sv = sinb[t * 32 + lane];

  for (int h = wave; h < kH; h += 8) {
    const float* qp = qraw + (size_t)row * kC + h * kD;
    const float x1 = qp[lane], x2 = qp[lane + 32];
    const float r1 =  x1 * cv + x2 * sv;
    const float r2 = -x1 * sv + x2 * cv;
    float ssq = r1 * r1 + r2 * r2;
#pragma unroll
    for (int off = 16; off > 0; off >>= 1) ssq += __shfl_xor(ssq, off);
    const float inv = rsqrtf(ssq * (1.0f / 64.0f) + 1e-6f) * 1.2f;
    qn[h][lane]      = r1 * inv;
    qn[h][lane + 32] = r2 * inv;
  }
  if (wave < kJ) {
    const int j = wave;
    const float* kp = kraw + (size_t)row * kJD + j * kD;
    const float x1 = kp[lane], x2 = kp[lane + 32];
    const float r1 =  x1 * cv + x2 * sv;
    const float r2 = -x1 * sv + x2 * cv;
    float ssq = r1 * r1 + r2 * r2;
#pragma unroll
    for (int off = 16; off > 0; off >>= 1) ssq += __shfl_xor(ssq, off);
    const float inv = rsqrtf(ssq * (1.0f / 64.0f) + 1e-6f) * 1.2f;
    kflat[(size_t)row * kJD + j * kD + lane]      = (_Float16)(r1 * inv);
    kflat[(size_t)row * kJD + j * kD + lane + 32] = (_Float16)(r2 * inv);
  }
  vbh[(size_t)row * kJD + tid] = (_Float16)vraw[(size_t)row * kJD + tid];
  __syncthreads();

  if (tid < kH) {
    const int h = tid;
    float lk[kJ], lv[kJ];
    float mk = -1e30f, mv = -1e30f;
#pragma unroll
    for (int j = 0; j < kJ; ++j) {
      lk[j] = akraw[(size_t)row * 64 + h * kJ + j] + bak[h * kJ + j];
      lv[j] = avraw[(size_t)row * 64 + h * kJ + j] + bav[h * kJ + j];
      mk = fmaxf(mk, lk[j]);
      mv = fmaxf(mv, lv[j]);
    }
    float sk = 0.f, sv2 = 0.f;
#pragma unroll
    for (int j = 0; j < kJ; ++j) {
      lk[j] = __expf(lk[j] - mk); sk += lk[j];
      lv[j] = __expf(lv[j] - mv); sv2 += lv[j];
    }
#pragma unroll
    for (int j = 0; j < kJ; ++j) {
      wks[h][j] = lk[j] / sk;
      wvw[(size_t)row * 64 + h * kJ + j] = lv[j] / sv2;
    }
    float g = 0.f;
#pragma unroll
    for (int c = 0; c < 12; ++c) g += x[(size_t)row * kC + c] * wgate[h * 12 + c];
    gateg[(size_t)row * kH + h] = 3.0f / (1.0f + __expf(-g));
  }
  __syncthreads();

  for (int idx = tid; idx < kH * kJD; idx += 256) {
    const int h = idx >> 8, jd = idx & 255, j = jd >> 6, d = jd & 63;
    const float val = wks[h][j] * qn[h][d] * 2.0f;
    qeff[((size_t)(b * kH + h) * kT + t) * kJD + jd] = (_Float16)val;
  }
}

// ---------------------------------------------------------------------------
// Flash attention per (b,h). 128 q rows/block, 8 waves x 16-row strips,
// s-tiles of 32. K-tile fetched by the Tensor Data Mover (with hardware
// padding to the 264-half LDS stride); V-tile combined (vb + gate*ve) by VALU.
// ---------------------------------------------------------------------------
#define ATT_TQ 128
#define ATT_TS 32
#define KV_STR 264  // padded jd stride
#define P_STR  40

__global__ void attn_kernel(const _Float16* __restrict__ qeff, const _Float16* __restrict__ kflat,
                            const _Float16* __restrict__ vbh, const _Float16* __restrict__ veh,
                            const float* __restrict__ gateg, const float* __restrict__ wvw,
                            float* __restrict__ ypre) {
  __shared__ _Float16 Ks[ATT_TS][KV_STR];
  __shared__ _Float16 Vs[ATT_TS][KV_STR];
  __shared__ _Float16 Ps[8][16][P_STR];

  const int bh = blockIdx.y;
  const int b = bh >> 4, h = bh & 15;
  const int q0 = blockIdx.x * ATT_TQ;
  const int tid = threadIdx.x, wave = tid >> 5, lane = tid & 31;
  const int n = lane & 15, hi = lane >> 4;
  const int qrow0 = q0 + wave * 16;

  const _Float16* qbase = qeff + ((size_t)(b * kH + h) * kT + qrow0) * kJD;
  v16h qa[8];
#pragma unroll
  for (int kt = 0; kt < 8; ++kt) qa[kt] = load_frag_a(qbase + kt * 32, kJD);

  v8f oacc[16];
#pragma unroll
  for (int i = 0; i < 16; ++i) { v8f z = {}; oacc[i] = z; }
  float mrow[8], lrow[8];
#pragma unroll
  for (int r = 0; r < 8; ++r) { mrow[r] = -1e30f; lrow[r] = 0.f; }

#if USE_TDM
  // Static half of the K-tile tensor descriptor (D#), per CDNA5 ISA 8.3/8.4.
  // 2D tensor: dim0 = 256 halves (one row), dim1 = B*T rows, stride0 = 256.
  // Tile: 256 x 32. data_size=2B. LDS padding: 4 DWORDs every 128 DWORDs
  // (pad_interval=6, pad_amount=3) -> 264-half row stride in LDS.
  const unsigned int ldsK = (unsigned int)(uintptr_t)&Ks[0][0];  // low 32 bits = LDS offset
  i32x8 g1;
  g1[0] = (1 << 16) | (1 << 20) | (6 << 22) | (3 << 25);
  g1[1] = (kJD & 0xFFFF) << 16;                 // tensor_dim0[15:0]
  g1[2] = (kJD >> 16) | ((kM & 0xFFFF) << 16);  // tensor_dim0[31:16] | tensor_dim1[15:0]
  g1[3] = (kM >> 16) | (kJD << 16);             // tensor_dim1[31:16] | tile_dim0
  g1[4] = ATT_TS;                               // tile_dim1
  g1[5] = kJD;                                  // tensor_dim0_stride[31:0]
  g1[6] = 0;
  g1[7] = (int)(((long long)kM * kJD) >> 16);   // tensor_dim1_stride[47:16]
  i32x4 gz4 = {};
  i32x8 gz8 = {};
#endif

  const int ldr = tid >> 3;            // 0..31 (s row within tile)
  const int ldc = (tid & 7) * 32;      // col base

  const int smax = q0 + ATT_TQ;        // causal: s tiles with s0 < smax
  for (int s0 = 0; s0 < smax; s0 += ATT_TS) {
#if USE_TDM
    if (wave == 0) {
      const unsigned long long ga =
          (unsigned long long)(uintptr_t)(kflat + (size_t)(b * kT + s0) * kJD);
      u32x4 g0;
      g0[0] = 1u;                       // count=1 (valid descriptor)
      g0[1] = ldsK;                     // lds_addr
      g0[2] = (unsigned int)ga;         // global_addr[31:0]
      g0[3] = (unsigned int)(ga >> 32) | (2u << 30);  // global_addr[56:32] | type=2
      __builtin_amdgcn_tensor_load_to_lds(g0, g1, gz4, gz4, gz8, 0);
    }
#endif
    {
      const size_t srow = (size_t)(b * kT + s0 + ldr) * kJD;
      const float g = gateg[(size_t)(b * kT + s0 + ldr) * kH + h];
#pragma unroll
      for (int c = 0; c < 32; c += 8) {
#if !USE_TDM
        *(v8h*)&Ks[ldr][ldc + c] = *(const v8h*)&kflat[srow + ldc + c];
#endif
        v8h vb = *(const v8h*)&vbh[srow + ldc + c];
        v8h ve = *(const v8h*)&veh[srow + ldc + c];
        v8h vc;
#pragma unroll
        for (int e = 0; e < 8; ++e) vc[e] = (_Float16)((float)vb[e] + g * (float)ve[e]);
        *(v8h*)&Vs[ldr][ldc + c] = vc;
      }
      if (s0 + ATT_TS < smax) {  // prefetch next s-tile (global_prefetch_b8)
        const size_t srow2 = srow + (size_t)ATT_TS * kJD;
        __builtin_prefetch((const void*)(vbh + srow2 + ldc), 0, 0);
        __builtin_prefetch((const void*)(veh + srow2 + ldc), 0, 0);
      }
    }
#if USE_TDM
    if (wave == 0) __builtin_amdgcn_s_wait_tensorcnt(0);
#endif
    __syncthreads();

    if (s0 <= qrow0 + 15) {
      v8f sacc[2];
#pragma unroll
      for (int i = 0; i < 2; ++i) { v8f z = {}; sacc[i] = z; }
#pragma unroll
      for (int kt = 0; kt < 8; ++kt) {
#pragma unroll
        for (int nt = 0; nt < 2; ++nt) {
          v16h bf = load_frag_b(&Ks[nt * 16][kt * 32], 1, KV_STR);
          sacc[nt] = WMMA_F16(qa[kt], bf, sacc[nt]);
        }
      }
#pragma unroll
      for (int r = 0; r < 8; ++r) {
        const int tg = qrow0 + r + 8 * hi;
        float v2[2];
        float mx = -1e30f;
#pragma unroll
        for (int nt = 0; nt < 2; ++nt) {
          const int sg = s0 + nt * 16 + n;
          float v = sacc[nt][r] * 0.0625f;  // 1/sqrt(J*D)
          if (sg > tg) v = -1e30f;
          v2[nt] = v;
          mx = fmaxf(mx, v);
        }
#pragma unroll
        for (int off = 1; off < 16; off <<= 1) mx = fmaxf(mx, __shfl_xor(mx, off));
        const float newm = fmaxf(mrow[r], mx);
        const float corr = __expf(mrow[r] - newm);
        float rsum = 0.f;
#pragma unroll
        for (int nt = 0; nt < 2; ++nt) {
          const float pv = __expf(v2[nt] - newm);
          rsum += pv;
          Ps[wave][r + 8 * hi][nt * 16 + n] = (_Float16)pv;
        }
#pragma unroll
        for (int off = 1; off < 16; off <<= 1) rsum += __shfl_xor(rsum, off);
        lrow[r] = lrow[r] * corr + rsum;
        mrow[r] = newm;
#pragma unroll
        for (int i = 0; i < 16; ++i) oacc[i][r] *= corr;
      }
      v16h pa = load_frag_a(&Ps[wave][0][0], P_STR);
#pragma unroll
      for (int nt = 0; nt < 16; ++nt) {
        v16h bf = load_frag_b(&Vs[0][nt * 16], KV_STR, 1);
        oacc[nt] = WMMA_F16(pa, bf, oacc[nt]);
      }
    }
    __syncthreads();
  }

#pragma unroll
  for (int r = 0; r < 8; ++r) {
    const int m = r + 8 * hi;
    const int tg = qrow0 + m;
    const float invl = 1.0f / lrow[r];
    const float* wv = wvw + ((size_t)(b * kT + tg) * kH + h) * kJ;
    const float w0 = wv[0], w1 = wv[1], w2 = wv[2], w3 = wv[3];
#pragma unroll
    for (int dg = 0; dg < 4; ++dg) {
      const float val = w0 * oacc[dg][r] + w1 * oacc[dg + 4][r] +
                        w2 * oacc[dg + 8][r] + w3 * oacc[dg + 12][r];
      const int d = dg * 16 + n;
      ypre[((size_t)(b * kT + tg) * kH + h) * kD + d] = val * invl;
    }
  }
}

// ---------------------------------------------------------------------------
// Host-side launch
// ---------------------------------------------------------------------------
extern "C" void kernel_launch(void* const* d_in, const int* in_sizes, int n_in,
                              void* d_out, int out_size, void* d_ws, size_t ws_size,
                              hipStream_t stream) {
  const float* x     = (const float*)d_in[0];
  const float* ve    = (const float*)d_in[1];
  const float* cosb  = (const float*)d_in[2];
  const float* sinb  = (const float*)d_in[3];
  const float* wq    = (const float*)d_in[4];
  const float* wk    = (const float*)d_in[5];
  const float* wv    = (const float*)d_in[6];
  const float* wproj = (const float*)d_in[7];
  const float* wak   = (const float*)d_in[8];
  const float* wav   = (const float*)d_in[9];
  const float* bak   = (const float*)d_in[10];
  const float* bav   = (const float*)d_in[11];
  const float* wgate = (const float*)d_in[12];
  float* out = (float*)d_out;

  uintptr_t p = (uintptr_t)d_ws;
  auto alloc = [&](size_t bytes) -> void* {
    void* r = (void*)p;
    p += (bytes + 255) & ~(size_t)255;
    return r;
  };
  _Float16* xh     = (_Float16*)alloc((size_t)kM * kC * 2);
  _Float16* wqh    = (_Float16*)alloc((size_t)kC * kC * 2);
  _Float16* wkh    = (_Float16*)alloc((size_t)kJD * kC * 2);
  _Float16* wvh    = (_Float16*)alloc((size_t)kJD * kC * 2);
  _Float16* wakh   = (_Float16*)alloc((size_t)64 * kC * 2);
  _Float16* wavh   = (_Float16*)alloc((size_t)64 * kC * 2);
  _Float16* wprojh = (_Float16*)alloc((size_t)kC * kC * 2);
  float*    qraw   = (float*)alloc((size_t)kM * kC * 4);
  float*    kraw   = (float*)alloc((size_t)kM * kJD * 4);
  float*    vraw   = (float*)alloc((size_t)kM * kJD * 4);
  float*    akraw  = (float*)alloc((size_t)kM * 64 * 4);
  float*    avraw  = (float*)alloc((size_t)kM * 64 * 4);
  _Float16* qeff   = (_Float16*)alloc((size_t)kB * kH * kT * kJD * 2);
  _Float16* kflat  = (_Float16*)alloc((size_t)kM * kJD * 2);
  _Float16* vbh    = (_Float16*)alloc((size_t)kM * kJD * 2);
  _Float16* veh    = (_Float16*)alloc((size_t)kM * kJD * 2);
  float*    gateg  = (float*)alloc((size_t)kM * kH * 4);
  float*    wvw    = (float*)alloc((size_t)kM * 64 * 4);
  float*    ypre   = (float*)alloc((size_t)kM * kC * 4);
  _Float16* ypreh  = (_Float16*)alloc((size_t)kM * kC * 2);

  auto cvt = [&](const float* s, _Float16* d, int nel) {
    cvt_f32_f16<<<dim3((nel + 255) / 256), dim3(256), 0, stream>>>(s, d, nel);
  };
  cvt(x, xh, kM * kC);
  cvt(ve, veh, kM * kJD);
  cvt(wq, wqh, kC * kC);
  cvt(wk, wkh, kJD * kC);
  cvt(wv, wvh, kJD * kC);
  cvt(wak, wakh, 64 * kC);
  cvt(wav, wavh, 64 * kC);
  cvt(wproj, wprojh, kC * kC);

  const dim3 blk(256);
  gemm_f16f32<<<dim3(kM / 64, kC / 64), blk, 0, stream>>>(xh, wqh, qraw, kM, kC, kC);
  gemm_f16f32<<<dim3(kM / 64, kJD / 64), blk, 0, stream>>>(xh, wkh, kraw, kM, kJD, kC);
  gemm_f16f32<<<dim3(kM / 64, kJD / 64), blk, 0, stream>>>(xh, wvh, vraw, kM, kJD, kC);
  gemm_f16f32<<<dim3(kM / 64, 64 / 64), blk, 0, stream>>>(xh, wakh, akraw, kM, 64, kC);
  gemm_f16f32<<<dim3(kM / 64, 64 / 64), blk, 0, stream>>>(xh, wavh, avraw, kM, 64, kC);

  prep_kernel<<<dim3(kM), blk, 0, stream>>>(qraw, kraw, vraw, akraw, avraw, x, cosb, sinb,
                                            bak, bav, wgate, qeff, kflat, vbh, wvw, gateg);

  attn_kernel<<<dim3(kT / ATT_TQ, kB * kH), blk, 0, stream>>>(qeff, kflat, vbh, veh, gateg,
                                                              wvw, ypre);

  cvt(ypre, ypreh, kM * kC);
  gemm_f16f32<<<dim3(kM / 64, kC / 64), blk, 0, stream>>>(ypreh, wprojh, out, kM, kC, kC);
}
